// GGNN_28028956574231
// MI455X (gfx1250) — matrix operand order repaired and checked by reading, not compiled, not run on hardware
//
#include <hip/hip_runtime.h>
#include <hip/hip_bf16.h>
#include <math.h>

// ---------------------------------------------------------------------------
// GGNN: 5 layers of { m = h@W[i] ; agg = scatter_add(ew * m[src] -> dst) ;
//                     h = GRUCell(agg, h) } ; out = relu(h) @ fc_w.T + fc_b
// H = 256. GEMMs: v_wmma_f32_16x16x32_bf16 (fp32 accumulate).
// Weight panels are staged to LDS via the Tensor Data Mover (TENSORcnt).
// ---------------------------------------------------------------------------

typedef __attribute__((ext_vector_type(16))) __bf16        v16bf;
typedef __attribute__((ext_vector_type(8)))  float         v8f;
typedef __attribute__((ext_vector_type(4)))  unsigned int  v4u;
typedef __attribute__((ext_vector_type(8)))  int           v8i;
typedef __attribute__((ext_vector_type(4)))  int           v4i;

#define Hc 256

// ---------------------------------------------------------------------------
// TDM: async 2D-tile load Global -> LDS (ISA 08_async_tensor 7/8).
//   group0: count=1 | lds_addr | global_addr[56:0] | type=2
//   group1: data_size=1B, tensor_dim0=rowBytes, tensor_dim1=nRows,
//           tile_dim0=rowBytes, tile_dim1=nRows, tensor_dim0_stride=strideBytes
// Issued by one wave per block; completion via s_wait_tensorcnt + barrier.
// ---------------------------------------------------------------------------
__device__ __forceinline__ void tdm_load_2d_to_lds(const void* gsrc, unsigned ldsByteOff,
                                                   unsigned rowBytes, unsigned nRows,
                                                   unsigned strideBytes) {
  const unsigned long long ga = (unsigned long long)gsrc;
  v4u g0;
  g0[0] = 1u;                                              // count=1, user mode
  g0[1] = ldsByteOff;                                      // lds_addr
  g0[2] = (unsigned)(ga & 0xFFFFFFFFu);                    // global_addr[31:0]
  g0[3] = (unsigned)((ga >> 32) & 0x01FFFFFFu) | 0x80000000u; // [56:32] | type=2

  v8i g1;
  g1[0] = 0;                                               // wg_mask=0, data_size=0 (1B)
  g1[1] = (int)((rowBytes & 0xFFFFu) << 16);               // tensor_dim0[15:0] @bit48
  g1[2] = (int)(((rowBytes >> 16) & 0xFFFFu) |             // tensor_dim0[31:16]
                ((nRows & 0xFFFFu) << 16));                // tensor_dim1[15:0]
  g1[3] = (int)(((nRows >> 16) & 0xFFFFu) |                // tensor_dim1[31:16]
                ((rowBytes & 0xFFFFu) << 16));             // tile_dim0
  g1[4] = (int)(nRows & 0xFFFFu);                          // tile_dim1 (tile_dim2=0)
  g1[5] = (int)strideBytes;                                // tensor_dim0_stride[31:0]
  g1[6] = 0;                                               // stride hi / dim1_stride lo
  g1[7] = 0;

  v4i z4 = {0, 0, 0, 0};
#if defined(__clang_major__) && (__clang_major__ >= 23)
  v8i z8 = {};
  __builtin_amdgcn_tensor_load_to_lds(g0, g1, z4, z4, z8, 0);
#else
  __builtin_amdgcn_tensor_load_to_lds(g0, g1, z4, z4, 0);
#endif
}

__device__ __forceinline__ unsigned lds_byte_offset(const void* p) {
  // LDS aperture: addr[31:0] is the LDS byte offset (ISA 10.2 aperture calc).
  return (unsigned)(unsigned long long)p;
}

// ---------------------------------------------------------------------------
// A-fragment loader: 16x32 bf16 A-tile per ISA 7.12.2 layout, built from a
// row-major fp32 matrix (converted in-register with v_cvt_pk_bf16_f32).
//   lanes 0-15 : row = rowBase+lane,    K = kk+0..7,  kk+16..23
//   lanes16-31 : row = rowBase+lane-16, K = kk+8..15, kk+24..31
// ---------------------------------------------------------------------------
__device__ __forceinline__ v16bf load_a_frag_f32(const float* __restrict__ A,
                                                 int ldA, int rowBase, int Mrows,
                                                 int kk, int lane) {
  int r = rowBase + (lane & 15);
  if (r >= Mrows) r = Mrows - 1;                 // clamp; keeps EXEC all-ones
  const float* p = A + (size_t)r * ldA + kk + ((lane >> 4) << 3);
  v16bf a;
#pragma unroll
  for (int i = 0; i < 8; ++i) {
    a[i]     = (__bf16)p[i];
    a[8 + i] = (__bf16)p[16 + i];
  }
  return a;
}

// B fragments pre-packed so each lane reads ONE contiguous 32B chunk.
// Packed order: [kt][nt][lane][i], element (lane,i) =
//   B[ kt*32 + 8*(lane>>4) + (i&7) + 16*(i>>3) ][ nt*16 + (lane&15) ]
__device__ __forceinline__ v16bf load_b_frag_lds(const __bf16* lds, int slot, int lane) {
  return *(const v16bf*)(lds + ((size_t)slot * 32 + lane) * 16);
}

__device__ __forceinline__ float sigmoidf_(float x) {
  return 1.0f / (1.0f + __expf(-x));
}

// ---------------------------------------------------------------------------
// Repack fp32 weight matrix into bf16 WMMA B-fragment order.
// transpose==0 : B[k][n] = src[k*Nc + n]   (weight[i], already [K,N])
// transpose==1 : B[k][n] = src[n*K + k]    (w_ih / w_hh, stored [N,K])
// ---------------------------------------------------------------------------
__global__ void repack_w_kernel(const float* __restrict__ src, __bf16* __restrict__ dst,
                                int K, int Nc, int transpose, int batch) {
  const int KT = K / 32, NT = Nc / 16;
  const size_t per = (size_t)KT * NT * 512;
  size_t idx = (size_t)blockIdx.x * blockDim.x + threadIdx.x;
  if (idx >= per * batch) return;
  const int    b    = (int)(idx / per);
  const size_t r    = idx % per;
  const int    i    = (int)(r & 15);
  const int    lane = (int)((r >> 4) & 31);
  const int    ntk  = (int)(r >> 9);
  const int    nt   = ntk % NT;
  const int    kt   = ntk / NT;
  const int    col  = nt * 16 + (lane & 15);
  const int    k    = kt * 32 + ((lane >> 4) << 3) + (i & 7) + ((i >> 3) << 4);
  const float* s    = src + (size_t)b * K * Nc;
  const float  v    = transpose ? s[(size_t)col * K + k] : s[(size_t)k * Nc + col];
  dst[idx] = (__bf16)v;
}

__global__ void zero_kernel(float4* __restrict__ p, size_t n4) {
  size_t i = (size_t)blockIdx.x * blockDim.x + threadIdx.x;
  if (i < n4) p[i] = make_float4(0.f, 0.f, 0.f, 0.f);
}

// ---------------------------------------------------------------------------
// m = h @ W[i] : [M,256] x [256,256]. Block = 128 threads (4 waves).
// Block tile 64x64; wave tile 16x64 (4 WMMAs per K-step).
// B panel (8 kt x 4 nt x 512 frag-elems = 32KB) staged once per block by TDM.
// ---------------------------------------------------------------------------
__global__ void gemm_m_kernel(const float* __restrict__ A, const __bf16* __restrict__ Bp,
                              float* __restrict__ C, int Mrows) {
  const int lane    = threadIdx.x & 31;
  const int wave    = threadIdx.x >> 5;
  const int rowBase = blockIdx.x * 64 + wave * 16;
  const int colBase = blockIdx.y * 64;
  const int NT      = Hc / 16;

  __shared__ __align__(64) __bf16 ldsB[8 * 4 * 512];        // 32 KB

  if (threadIdx.x < 32) {
    // rows = kt (8), row = 4 consecutive nt panels (4KB), row stride = NT*1KB
    const __bf16* src = Bp + (size_t)(colBase >> 4) * 512;
    tdm_load_2d_to_lds(src, lds_byte_offset(ldsB),
                       /*rowBytes=*/4 * 512 * 2, /*nRows=*/8,
                       /*strideBytes=*/NT * 512 * 2);
    __builtin_amdgcn_s_wait_tensorcnt(0);
  }
  __syncthreads();

  v8f acc[4] = {};
#pragma unroll
  for (int kt = 0; kt < Hc / 32; ++kt) {
    v16bf a = load_a_frag_f32(A, Hc, rowBase, Mrows, kt * 32, lane);
#pragma unroll
    for (int nt = 0; nt < 4; ++nt) {
      v16bf b = load_b_frag_lds(ldsB, kt * 4 + nt, lane);
      acc[nt] = __builtin_amdgcn_wmma_f32_16x16x32_bf16(false, a, false, b,
                                                        (short)0, acc[nt], false, false);
    }
  }

  const int col0 = colBase + (lane & 15);
  const int half = lane >> 4;
#pragma unroll
  for (int nt = 0; nt < 4; ++nt) {
#pragma unroll
    for (int r = 0; r < 8; ++r) {
      const int row = rowBase + r + 8 * half;
      if (row < Mrows) C[(size_t)row * Hc + col0 + nt * 16] = acc[nt][r];
    }
  }
}

// ---------------------------------------------------------------------------
// Edge scatter: agg[dst] += ew[e] * m[src].  64 lanes x float4 per edge.
// unsafeAtomicAdd -> hardware global_atomic_add_f32 (no return, STOREcnt);
// m and agg (51MB each) are L2-resident on the 192MB L2.
// ---------------------------------------------------------------------------
__global__ void scatter_kernel(const float* __restrict__ m, const int* __restrict__ ei,
                               const float* __restrict__ ew, float* __restrict__ agg,
                               int E) {
  const long long t = (long long)blockIdx.x * blockDim.x + threadIdx.x;
  const int e = (int)(t >> 6);
  if (e >= E) return;
  const int c = (int)(t & 63) << 2;
  const int s = ei[e];
  const int d = ei[(size_t)E + e];
  const float w = ew[e];
  const float4 v = *(const float4*)(m + (size_t)s * Hc + c);
  float* o = agg + (size_t)d * Hc + c;
  unsafeAtomicAdd(o + 0, w * v.x);
  unsafeAtomicAdd(o + 1, w * v.y);
  unsafeAtomicAdd(o + 2, w * v.z);
  unsafeAtomicAdd(o + 3, w * v.w);
}

// ---------------------------------------------------------------------------
// Fused GRU: gi = agg@w_ihT and gh = h@w_hhT computed tile-wise with 6 WMMA
// accumulators per wave (r/z/n per operand); gates applied in-register so
// gi/gh (2 x 153MB) are never materialized. Block = 4 waves; wave tile =
// 16 rows x 16 gate-columns. Six 8KB B panels (r/z/n of both weights) are
// TDM-staged to LDS (48KB) once per block.
// ---------------------------------------------------------------------------
__global__ void fused_gru_kernel(const float* __restrict__ agg, const float* __restrict__ h,
                                 const __bf16* __restrict__ WihTp, const __bf16* __restrict__ WhhTp,
                                 const float* __restrict__ b_ih, const float* __restrict__ b_hh,
                                 float* __restrict__ h_out, int Mrows) {
  const int lane    = threadIdx.x & 31;
  const int wave    = threadIdx.x >> 5;
  const int rowBase = blockIdx.x * 64 + wave * 16;
  const int nt      = blockIdx.y;       // 0..15 (column strip j = nt*16)
  const int NT      = 768 / 16;         // 48

  __shared__ __align__(64) __bf16 ldsG[6 * 8 * 512];        // 48 KB

  if (threadIdx.x < 32) {
#pragma unroll
    for (int p = 0; p < 6; ++p) {
      const __bf16* W = (p < 3) ? WihTp : WhhTp;
      const int ntp = nt + 16 * (p % 3);                    // gate column block
      tdm_load_2d_to_lds(W + (size_t)ntp * 512,
                         lds_byte_offset(ldsG) + (unsigned)p * 8 * 512 * 2,
                         /*rowBytes=*/512 * 2, /*nRows=*/8,
                         /*strideBytes=*/NT * 512 * 2);
    }
    __builtin_amdgcn_s_wait_tensorcnt(0);
  }
  __syncthreads();

  v8f ir = {}, iz = {}, in_ = {}, hr = {}, hz = {}, hn = {};
#pragma unroll
  for (int kt = 0; kt < Hc / 32; ++kt) {
    v16bf aA = load_a_frag_f32(agg, Hc, rowBase, Mrows, kt * 32, lane);
    v16bf aH = load_a_frag_f32(h,   Hc, rowBase, Mrows, kt * 32, lane);

    v16bf b0 = load_b_frag_lds(ldsG, 0 * 8 + kt, lane);
    v16bf b1 = load_b_frag_lds(ldsG, 1 * 8 + kt, lane);
    v16bf b2 = load_b_frag_lds(ldsG, 2 * 8 + kt, lane);
    ir  = __builtin_amdgcn_wmma_f32_16x16x32_bf16(false, aA, false, b0, (short)0, ir,  false, false);
    iz  = __builtin_amdgcn_wmma_f32_16x16x32_bf16(false, aA, false, b1, (short)0, iz,  false, false);
    in_ = __builtin_amdgcn_wmma_f32_16x16x32_bf16(false, aA, false, b2, (short)0, in_, false, false);

    v16bf c0 = load_b_frag_lds(ldsG, 3 * 8 + kt, lane);
    v16bf c1 = load_b_frag_lds(ldsG, 4 * 8 + kt, lane);
    v16bf c2 = load_b_frag_lds(ldsG, 5 * 8 + kt, lane);
    hr = __builtin_amdgcn_wmma_f32_16x16x32_bf16(false, aH, false, c0, (short)0, hr, false, false);
    hz = __builtin_amdgcn_wmma_f32_16x16x32_bf16(false, aH, false, c1, (short)0, hz, false, false);
    hn = __builtin_amdgcn_wmma_f32_16x16x32_bf16(false, aH, false, c2, (short)0, hn, false, false);
  }

  const int col  = nt * 16 + (lane & 15);
  const int half = lane >> 4;
  const float bir = b_ih[col], biz = b_ih[col + 256], bin = b_ih[col + 512];
  const float bhr = b_hh[col], bhz = b_hh[col + 256], bhn = b_hh[col + 512];

#pragma unroll
  for (int r = 0; r < 8; ++r) {
    const int row = rowBase + r + 8 * half;
    const int rc  = row < Mrows ? row : Mrows - 1;
    const float hold = h[(size_t)rc * Hc + col];
    const float rg = sigmoidf_(ir[r] + hr[r] + bir + bhr);
    const float zg = sigmoidf_(iz[r] + hz[r] + biz + bhz);
    const float ng = tanhf(in_[r] + bin + rg * (hn[r] + bhn));
    const float hv = (1.0f - zg) * ng + zg * hold;
    if (row < Mrows) h_out[(size_t)row * Hc + col] = hv;
  }
}

// ---------------------------------------------------------------------------
// out[n] = relu(h[n,:]) . fc_w + fc_b      (one wave32 per node)
// ---------------------------------------------------------------------------
__global__ void final_fc_kernel(const float* __restrict__ h, const float* __restrict__ fc_w,
                                const float* __restrict__ fc_b, float* __restrict__ out,
                                int Mrows) {
  const int lane = threadIdx.x & 31;
  const int wave = threadIdx.x >> 5;
  const int n    = blockIdx.x * 4 + wave;
  if (n >= Mrows) return;
  const float* hp = h + (size_t)n * Hc + lane * 8;
  const float* wp = fc_w + lane * 8;
  float s = 0.f;
#pragma unroll
  for (int i = 0; i < 8; ++i) {
    float v = hp[i];
    v = v > 0.f ? v : 0.f;
    s += v * wp[i];
  }
#pragma unroll
  for (int off = 16; off > 0; off >>= 1) s += __shfl_down(s, off, 32);
  if (lane == 0) out[n] = s + fc_b[0];
}

// ---------------------------------------------------------------------------
extern "C" void kernel_launch(void* const* d_in, const int* in_sizes, int n_in,
                              void* d_out, int out_size, void* d_ws, size_t ws_size,
                              hipStream_t stream) {
  const float* x      = (const float*)d_in[0];
  const int*   ei     = (const int*)d_in[1];   // [2,E] int32
  const float* ew     = (const float*)d_in[2];
  const float* weight = (const float*)d_in[3]; // [L,256,256]
  const float* w_ih   = (const float*)d_in[4]; // [768,256]
  const float* w_hh   = (const float*)d_in[5]; // [768,256]
  const float* b_ih   = (const float*)d_in[6];
  const float* b_hh   = (const float*)d_in[7];
  const float* fc_w   = (const float*)d_in[8];
  const float* fc_b   = (const float*)d_in[9];

  const int N = in_sizes[0] / Hc;                  // 50000
  const int E = in_sizes[2];                       // 800000
  const int L = in_sizes[3] / (Hc * Hc);           // 5

  const size_t nh = (size_t)N * Hc;
  float* h0  = (float*)d_ws;
  float* h1  = h0 + nh;
  float* agg = h1 + nh;
  float* m   = agg + nh;
  const size_t wP = (size_t)(Hc / 32) * (Hc / 16) * 512;     // 65536 / layer
  const size_t gP = (size_t)(Hc / 32) * (768 / 16) * 512;    // 196608
  __bf16* Wp    = (__bf16*)(m + nh);
  __bf16* WihTp = Wp + (size_t)L * wP;
  __bf16* WhhTp = WihTp + gP;

  hipMemcpyAsync(h0, x, nh * sizeof(float), hipMemcpyDeviceToDevice, stream);

  {
    const size_t tot = (size_t)L * wP;
    repack_w_kernel<<<(unsigned)((tot + 255) / 256), 256, 0, stream>>>(weight, Wp, Hc, Hc, 0, L);
    repack_w_kernel<<<(unsigned)((gP + 255) / 256), 256, 0, stream>>>(w_ih, WihTp, Hc, 768, 1, 1);
    repack_w_kernel<<<(unsigned)((gP + 255) / 256), 256, 0, stream>>>(w_hh, WhhTp, Hc, 768, 1, 1);
  }

  const int MB = (N + 63) / 64;
  float* hc = h0;
  float* hn = h1;
  for (int i = 0; i < L; ++i) {
    const size_t n4 = nh / 4;
    zero_kernel<<<(unsigned)((n4 + 255) / 256), 256, 0, stream>>>((float4*)agg, n4);

    gemm_m_kernel<<<dim3(MB, Hc / 64), 128, 0, stream>>>(hc, Wp + (size_t)i * wP, m, N);

    const long long st = (long long)E * 64;
    scatter_kernel<<<(unsigned)((st + 255) / 256), 256, 0, stream>>>(m, ei, ew, agg, E);

    fused_gru_kernel<<<dim3(MB, Hc / 16), 128, 0, stream>>>(agg, hc, WihTp, WhhTp,
                                                            b_ih, b_hh, hn, N);
    float* t = hc; hc = hn; hn = t;
  }

  final_fc_kernel<<<(N + 3) / 4, 128, 0, stream>>>(hc, fc_w, fc_b, (float*)d_out, N);
}
